// DeltaNet_31877247271567
// MI455X (gfx1250) — matrix operand order repaired and checked by reading, not compiled, not run on hardware
//
#include <hip/hip_runtime.h>
#include <cstdint>
#include <cstddef>

#define DEV __device__ __forceinline__

typedef __attribute__((ext_vector_type(16))) _Float16 v16h;
typedef __attribute__((ext_vector_type(8)))  float    v8f;

static constexpr int Bc = 2, Lc = 2048, Dc = 1024, Hc = 4;
static constexpr int DKc = 256, DVc = 256, CHc = 32;
static constexpr int Mrows = Bc * Lc;          // 4096 token rows
static constexpr int Grows = Mrows * Hc;       // 16384 gate rows
static constexpr int KGATE = 1056;             // 1040 padded to multiple of 32
static constexpr int DVS  = 64;                // dv slab width in delta kernel
static constexpr int NSLAB = DVc / DVS;        // 4

// ---------------- optional CDNA5 async global->LDS path ----------------
#ifndef ATHENA_ASYNC_LDS
#if defined(__has_builtin)
#if __has_builtin(__builtin_amdgcn_global_load_async_to_lds_b128) && \
    __has_builtin(__builtin_amdgcn_s_wait_asynccnt)
#define ATHENA_ASYNC_LDS 1
#endif
#endif
#endif
#ifndef ATHENA_ASYNC_LDS
#define ATHENA_ASYNC_LDS 0
#endif

#if ATHENA_ASYNC_LDS
typedef int v4i_vec __attribute__((vector_size(16)));
DEV void async_cp16(void* lds, const void* g) {
  __builtin_amdgcn_global_load_async_to_lds_b128(
      (__attribute__((address_space(1))) v4i_vec*)(g),
      (__attribute__((address_space(3))) v4i_vec*)(lds), 0, 0);
}
DEV void async_wait0() { __builtin_amdgcn_s_wait_asynccnt(0); }
#endif

// ---------------- WMMA fragment helpers (16x16x32 f16, wave32) ----------------
// A 16x32 f16 (MxK): lane<16 -> K 0..15 ; lane>=16 -> K 16..31
DEV v16h frag_a_f16(const _Float16* A, int lda, int m0, int k0, int lane) {
  v16h a;
  const _Float16* row = A + (size_t)(m0 + (lane & 15)) * lda;
  int kb0 = k0 + ((lane & 16) ? 16 : 0);
#pragma unroll
  for (int j = 0; j < 16; ++j) a[j] = row[kb0 + j];
  return a;
}
// B 32x16 f16 (KxN) row-major: lane<16 -> K 0..15 ; lane>=16 -> K 16..31
DEV v16h frag_b_f16(const _Float16* Bm, int ldb, int k0, int n0, int lane) {
  v16h b;
  int n = n0 + (lane & 15);
  int kk = k0 + ((lane & 16) ? 16 : 0);
#pragma unroll
  for (int j = 0; j < 16; ++j) b[j] = Bm[(size_t)(kk + j) * ldb + n];
  return b;
}
// B = K^T where K stored row-major [32 x ldk] (chunk-rows x dk): B(kd,n)=Ksm[n][kd]
DEV v16h frag_bT_f16(const _Float16* Km, int ldk, int k0, int n0, int lane) {
  v16h b;
  int n = n0 + (lane & 15);
  int kd = k0 + ((lane & 16) ? 16 : 0);
#pragma unroll
  for (int j = 0; j < 16; ++j) b[j] = Km[(size_t)n * ldk + kd + j];
  return b;
}
// A = K^T with K stored [32 x ldk]: A(m,kk)=Ksm[kk][m]  (K-dim = 32 chunk rows)
DEV v16h frag_aT_f16(const _Float16* Km, int ldk, int m0, int lane) {
  v16h a;
  int m = m0 + (lane & 15);
  int c0 = (lane & 16) ? 16 : 0;
#pragma unroll
  for (int j = 0; j < 16; ++j) a[j] = Km[(size_t)(c0 + j) * ldk + m];
  return a;
}
// B fragment sourced from f32 LDS (converted to f16 on load)
DEV v16h frag_b_f32l(const float* Bm, int ldb, int k0, int n0, int lane) {
  v16h b;
  int n = n0 + (lane & 15);
  int kk = k0 + ((lane & 16) ? 16 : 0);
#pragma unroll
  for (int j = 0; j < 16; ++j) b[j] = (_Float16)Bm[(size_t)(kk + j) * ldb + n];
  return b;
}
// C/D f32 16x16: VGPR r: lanes<16 -> M=r, lanes>=16 -> M=r+8
DEV v8f frag_c_f32l(const float* C, int ldc, int m0, int n0, int lane) {
  v8f c;
  int n = n0 + (lane & 15);
  int mb = m0 + ((lane & 16) ? 8 : 0);
#pragma unroll
  for (int r = 0; r < 8; ++r) c[r] = C[(size_t)(mb + r) * ldc + n];
  return c;
}
DEV void store_c_f32p(float* C, int ldc, int m0, int n0, int lane, v8f acc) {
  int n = n0 + (lane & 15);
  int mb = m0 + ((lane & 16) ? 8 : 0);
#pragma unroll
  for (int r = 0; r < 8; ++r) C[(size_t)(mb + r) * ldc + n] = acc[r];
}
DEV void store_c_f16p(_Float16* C, int ldc, int m0, int n0, int lane, v8f acc) {
  int n = n0 + (lane & 15);
  int mb = m0 + ((lane & 16) ? 8 : 0);
#pragma unroll
  for (int r = 0; r < 8; ++r) C[(size_t)(mb + r) * ldc + n] = (_Float16)acc[r];
}

#define WMMA_F16(a, b, c) \
  __builtin_amdgcn_wmma_f32_16x16x32_f16(false, (a), false, (b), (short)0, (c), false, false)

// ---------------- small helpers ----------------
DEV float block_sum(float v, float* red) {
  int t = threadIdx.x;
  red[t] = v; __syncthreads();
  for (int s = 128; s > 0; s >>= 1) { if (t < s) red[t] += red[t + s]; __syncthreads(); }
  float r = red[0]; __syncthreads();
  return r;
}
DEV float sigmoidf_(float x) { return 1.f / (1.f + __expf(-x)); }

// ---------------- conversion kernels ----------------
__global__ void cvt_f16_kernel(const float* __restrict__ src, _Float16* __restrict__ dst, int n) {
  int i = blockIdx.x * 256 + threadIdx.x;
  if (i < n) dst[i] = (_Float16)src[i];
}
__global__ void cvt_pad_kernel(const float* __restrict__ src, _Float16* __restrict__ dst,
                               int rows_src, int rows_dst, int cols) {
  int i = blockIdx.x * 256 + threadIdx.x;
  if (i >= rows_dst * cols) return;
  int r = i / cols, c = i % cols;
  dst[i] = (r < rows_src) ? (_Float16)src[(size_t)r * cols + c] : (_Float16)0.f;
}

// ---------------- generic WMMA GEMM: C[MxN] = A[MxK](f16) * B[KxN](f16) ----------------
// EPI=0: store f32.  EPI=1: +bias, exact GELU, store f16.
template <int EPI>
__global__ void __launch_bounds__(128) wmma_gemm_kernel(
    const _Float16* __restrict__ A, const _Float16* __restrict__ Bm,
    void* __restrict__ Cout, const float* __restrict__ bias,
    int M, int N, int K, int lda, int ldb, int ldc) {
  int lane = threadIdx.x & 31;
  int wave = threadIdx.x >> 5;
  int ntn = N >> 4;
  int tile = blockIdx.x * 4 + wave;
  int tm = (tile / ntn) * 16;
  int tn = (tile % ntn) * 16;
  if (tm >= M) return;
  v8f acc = {};
  for (int k0 = 0; k0 < K; k0 += 32) {
    v16h a = frag_a_f16(A, lda, tm, k0, lane);
    v16h b = frag_b_f16(Bm, ldb, k0, tn, lane);
    if (k0 + 32 < K) {  // global_prefetch_b8 hints for next K-step
      __builtin_prefetch(A + (size_t)(tm + (lane & 15)) * lda + k0 + 32, 0, 3);
      __builtin_prefetch(Bm + (size_t)(k0 + 32 + (lane & 15)) * ldb + tn, 0, 3);
    }
    acc = WMMA_F16(a, b, acc);
  }
  int n = tn + (lane & 15);
  int mb = tm + ((lane & 16) ? 8 : 0);
  if constexpr (EPI == 0) {
    float* C = (float*)Cout;
#pragma unroll
    for (int r = 0; r < 8; ++r) C[(size_t)(mb + r) * ldc + n] = acc[r];
  } else {
    _Float16* C = (_Float16*)Cout;
    float bv = bias ? bias[n] : 0.f;
#pragma unroll
    for (int r = 0; r < 8; ++r) {
      float xg = acc[r] + bv;
      float g = 0.5f * xg * (1.f + erff(xg * 0.70710678118654752f));
      C[(size_t)(mb + r) * ldc + n] = (_Float16)g;
    }
  }
}

// ---------------- beta = sigmoid(x @ Wb) ----------------
__global__ void beta_kernel(const float* __restrict__ x, const float* __restrict__ Wb,
                            float* __restrict__ betab) {
  int idx = blockIdx.x * 256 + threadIdx.x;  // (b*L+l)*H + h
  if (idx >= Mrows * Hc) return;
  int h = idx & 3, row = idx >> 2;
  const float* xr = x + (size_t)row * Dc;
  float s = 0.f;
  for (int d = 0; d < Dc; ++d) s += xr[d] * Wb[d * Hc + h];
  betab[idx] = sigmoidf_(s);
}

// ---------------- dwconv(K=4)+SiLU, L2norm(q,k), beta-scale; emit f16 (b,h,l,d) ----------------
__global__ void __launch_bounds__(256) conv_qkv_kernel(
    const float* __restrict__ qp, const float* __restrict__ kp, const float* __restrict__ vp,
    const float* __restrict__ cq, const float* __restrict__ ck, const float* __restrict__ cv,
    const float* __restrict__ betab,
    _Float16* __restrict__ qn4, _Float16* __restrict__ kn4,
    _Float16* __restrict__ kb4, _Float16* __restrict__ vb4,
    float* __restrict__ vblhd) {
  __shared__ float red[256];
  int blk = blockIdx.x;             // (b*L+l)*H + h  == row16k
  int h = blk & (Hc - 1);
  int row = blk >> 2;               // b*L + l
  int b = row / Lc, l = row % Lc;
  int c = threadIdx.x;
  int ch = h * DKc + c;
  float qv = 0.f, kv = 0.f, vv = 0.f;
#pragma unroll
  for (int t = 0; t < 4; ++t) {
    int ls = l - 3 + t;
    if (ls >= 0) {
      size_t off = ((size_t)b * Lc + ls) * Dc + ch;
      qv += cq[ch * 4 + t] * qp[off];
      kv += ck[ch * 4 + t] * kp[off];
      vv += cv[ch * 4 + t] * vp[off];
    }
  }
  qv *= sigmoidf_(qv);  kv *= sigmoidf_(kv);  vv *= sigmoidf_(vv);
  float sq = block_sum(qv * qv, red);
  float sk = block_sum(kv * kv, red);
  float qn = qv * rsqrtf(sq + 1e-6f);
  float kn = kv * rsqrtf(sk + 1e-6f);
  float beta = betab[blk];
  size_t oidx = (((size_t)b * Hc + h) * Lc + l) * DKc + c;
  qn4[oidx] = (_Float16)qn;
  kn4[oidx] = (_Float16)kn;
  kb4[oidx] = (_Float16)(kn * beta);
  vb4[oidx] = (_Float16)(vv * beta);
  vblhd[(size_t)blk * DVc + c] = vv;       // (b,l,h,d)
}

// ---------------- chunkwise delta rule, S in f32 LDS, dv slabbed ----------------
static constexpr int SST  = DVS + 1;   // 65
static constexpr int QST  = DKc + 8;   // 264
static constexpr int VST  = DVS + 8;   // 72
static constexpr int AST  = CHc + 1;   // 33
static constexpr int AHST = 40;
static constexpr size_t DELTA_LDS =
    (size_t)DKc * SST * 4 + 4ull * CHc * QST * 2 + 2ull * CHc * VST * 2 +
    2ull * CHc * AHST * 2 + 2ull * CHc * AST * 4;   // ~153 KB

extern __shared__ char smem_raw[];

__global__ void __launch_bounds__(256) delta_kernel(
    const _Float16* __restrict__ qn4, const _Float16* __restrict__ kn4,
    const _Float16* __restrict__ kb4, const _Float16* __restrict__ vb4,
    float* __restrict__ delta_out) {
  float*    Ssm  = (float*)smem_raw;               // [DK][SST]
  _Float16* qsm  = (_Float16*)(Ssm + DKc * SST);   // [32][QST]
  _Float16* ksm  = qsm  + CHc * QST;
  _Float16* kbsm = ksm  + CHc * QST;
  _Float16* wsm  = kbsm + CHc * QST;
  _Float16* vbsm = wsm  + CHc * QST;               // [32][VST]
  _Float16* usm  = vbsm + CHc * VST;               // u then u_t
  _Float16* alh  = usm  + CHc * VST;               // [32][AHST]
  _Float16* aih  = alh  + CHc * AHST;              // Ainv f16
  float*    Asm  = (float*)(aih + CHc * AHST);     // [32][AST]
  float*    Aism = Asm + CHc * AST;

  int tid = threadIdx.x, lane = tid & 31, wave = tid >> 5;
  int wg = blockIdx.x;              // B*H*NSLAB workgroups
  int slab = wg & (NSLAB - 1);
  int bh = wg >> 2;                 // b*H + h
  int b = bh >> 2, h = bh & 3;

  for (int i = tid; i < DKc * SST; i += 256) Ssm[i] = 0.f;
  __syncthreads();

  const _Float16* qg0  = qn4 + (size_t)bh * Lc * DKc;
  const _Float16* kg0  = kn4 + (size_t)bh * Lc * DKc;
  const _Float16* kbg0 = kb4 + (size_t)bh * Lc * DKc;
  const _Float16* vbg0 = vb4 + (size_t)bh * Lc * DVc + slab * DVS;

  for (int chunk = 0; chunk < Lc / CHc; ++chunk) {
    const _Float16* qg  = qg0  + (size_t)chunk * CHc * DKc;
    const _Float16* kg  = kg0  + (size_t)chunk * CHc * DKc;
    const _Float16* kbg = kbg0 + (size_t)chunk * CHc * DKc;
    const _Float16* vbg = vbg0 + (size_t)chunk * CHc * DVc;

#if ATHENA_ASYNC_LDS
    for (int i = tid; i < CHc * 32; i += 256) {        // 32 x 16B segs per row
      int r = i >> 5, s = i & 31;
      async_cp16(qsm  + r * QST + s * 8, qg  + (size_t)r * DKc + s * 8);
      async_cp16(ksm  + r * QST + s * 8, kg  + (size_t)r * DKc + s * 8);
      async_cp16(kbsm + r * QST + s * 8, kbg + (size_t)r * DKc + s * 8);
    }
    for (int i = tid; i < CHc * 8; i += 256) {
      int r = i >> 3, s = i & 7;
      async_cp16(vbsm + r * VST + s * 8, vbg + (size_t)r * DVc + s * 8);
    }
    async_wait0();
#else
    for (int i = tid; i < CHc * DKc; i += 256) {
      int r = i >> 8, c = i & 255;
      qsm [r * QST + c] = qg [(size_t)r * DKc + c];
      ksm [r * QST + c] = kg [(size_t)r * DKc + c];
      kbsm[r * QST + c] = kbg[(size_t)r * DKc + c];
    }
    for (int i = tid; i < CHc * DVS; i += 256) {
      int r = i / DVS, c = i % DVS;
      vbsm[r * VST + c] = vbg[(size_t)r * DVc + c];
    }
#endif
    __syncthreads();

    // A = k_beta @ k^T  (32x32, K=256) : waves 0..3
    if (wave < 4) {
      int tm = (wave >> 1) * 16, tn = (wave & 1) * 16;
      v8f acc = {};
      for (int k0 = 0; k0 < DKc; k0 += 32) {
        v16h a  = frag_a_f16(kbsm, QST, tm, k0, lane);
        v16h bf = frag_bT_f16(ksm, QST, k0, tn, lane);
        acc = WMMA_F16(a, bf, acc);
      }
      store_c_f32p(Asm, AST, tm, tn, lane, acc);
    }
    __syncthreads();

    // attn_inv = (I + strict_lower(A))^{-1} via forward substitution (wave 0)
    if (tid < 32) {
      int j = tid;
      float v0 = (j == 0) ? 1.f : 0.f;
      Aism[j] = v0;
      aih[j] = (_Float16)v0;
      for (int r = 1; r < CHc; ++r) {
        float s = (j == r) ? 1.f : 0.f;
        for (int m = 0; m < r; ++m) s -= Asm[r * AST + m] * Aism[m * AST + j];
        s = (j <= r) ? s : 0.f;
        Aism[r * AST + j] = s;
        aih[r * AHST + j] = (_Float16)s;
      }
    }
    __syncthreads();

    // u = Ainv @ v_beta (32xDVS, one tile/wave) ; w = Ainv @ k_beta (32x256, 4 tiles/wave)
    {
      int tm = (wave >> 2) * 16, tn = (wave & 3) * 16;
      v16h a  = frag_a_f16(aih, AHST, tm, 0, lane);
      v16h bf = frag_b_f16(vbsm, VST, 0, tn, lane);
      v8f acc = {};
      acc = WMMA_F16(a, bf, acc);
      store_c_f16p(usm, VST, tm, tn, lane, acc);
    }
    for (int t = 0; t < 4; ++t) {
      int tile = wave * 4 + t;
      int tm = (tile >> 4) * 16, tn = (tile & 15) * 16;
      v16h a  = frag_a_f16(aih, AHST, tm, 0, lane);
      v16h bf = frag_b_f16(kbsm, QST, 0, tn, lane);
      v8f acc = {};
      acc = WMMA_F16(a, bf, acc);
      store_c_f16p(wsm, QST, tm, tn, lane, acc);
    }
    __syncthreads();

    // u_t = u - w @ S (one tile/wave)
    {
      int tm = (wave >> 2) * 16, tn = (wave & 3) * 16;
      v8f acc = {};
      for (int k0 = 0; k0 < DKc; k0 += 32) {
        v16h a  = frag_a_f16(wsm, QST, tm, k0, lane);
        v16h bf = frag_b_f32l(Ssm, SST, k0, tn, lane);
        acc = WMMA_F16(a, bf, acc);
      }
      int n = tn + (lane & 15);
      int mb = tm + ((lane & 16) ? 8 : 0);
      _Float16 ut[8];
#pragma unroll
      for (int r = 0; r < 8; ++r)
        ut[r] = (_Float16)((float)usm[(mb + r) * VST + n] - acc[r]);
#pragma unroll
      for (int r = 0; r < 8; ++r) usm[(mb + r) * VST + n] = ut[r];
    }
    __syncthreads();

    // attn_local = tril(q @ k^T) : waves 0..3
    if (wave < 4) {
      int tm = (wave >> 1) * 16, tn = (wave & 1) * 16;
      v8f acc = {};
      for (int k0 = 0; k0 < DKc; k0 += 32) {
        v16h a  = frag_a_f16(qsm, QST, tm, k0, lane);
        v16h bf = frag_bT_f16(ksm, QST, k0, tn, lane);
        acc = WMMA_F16(a, bf, acc);
      }
      int n = tn + (lane & 15);
      int mb = tm + ((lane & 16) ? 8 : 0);
#pragma unroll
      for (int r = 0; r < 8; ++r)
        alh[(mb + r) * AHST + n] = (_Float16)(((mb + r) >= n) ? acc[r] : 0.f);
    }
    __syncthreads();

    // o = q @ S + attn_local @ u_t  (one tile/wave) -> global (b,l,h,d)
    {
      int tm = (wave >> 2) * 16, tn = (wave & 3) * 16;
      v8f acc = {};
      for (int k0 = 0; k0 < DKc; k0 += 32) {
        v16h a  = frag_a_f16(qsm, QST, tm, k0, lane);
        v16h bf = frag_b_f32l(Ssm, SST, k0, tn, lane);
        acc = WMMA_F16(a, bf, acc);
      }
      v16h a  = frag_a_f16(alh, AHST, tm, 0, lane);
      v16h bf = frag_b_f16(usm, VST, 0, tn, lane);
      acc = WMMA_F16(a, bf, acc);
      int n = tn + (lane & 15);
      int mb = tm + ((lane & 16) ? 8 : 0);
#pragma unroll
      for (int r = 0; r < 8; ++r) {
        int lrow = chunk * CHc + mb + r;
        delta_out[(((size_t)b * Lc + lrow) * Hc + h) * DVc + slab * DVS + n] = acc[r];
      }
    }
    __syncthreads();

    // S += k^T @ u_t  (256xDVS, 8 tiles/wave, f32 accumulate in LDS)
    for (int t = 0; t < 8; ++t) {
      int tile = wave * 8 + t;
      int tm = (tile >> 2) * 16, tn = (tile & 3) * 16;
      v16h a  = frag_aT_f16(ksm, QST, tm, lane);
      v16h bf = frag_b_f16(usm, VST, 0, tn, lane);
      v8f acc = frag_c_f32l(Ssm, SST, tm, tn, lane);
      acc = WMMA_F16(a, bf, acc);
      store_c_f32p(Ssm, SST, tm, tn, lane, acc);
    }
    __syncthreads();
  }
}

// ---------------- FIR K=5 and K=64 on v (b,l,h,d) ----------------
__global__ void fir_kernel(const float* __restrict__ v, const float* __restrict__ fir_short,
                           const float* __restrict__ fir_long,
                           float* __restrict__ fs, float* __restrict__ fl) {
  int idx = blockIdx.x * 256 + threadIdx.x;   // b:1 l:11 h:2 d:8 bits
  int d = idx & 255;
  int h = (idx >> 8) & 3;
  int l = (idx >> 10) & (Lc - 1);
  int b = idx >> 21;
  int hd = h * DVc + d;
  float accs = 0.f;
#pragma unroll
  for (int t = 0; t < 5; ++t) {
    int ls = l - 4 + t;
    if (ls >= 0) accs += fir_short[hd * 5 + t] * v[((((size_t)b * Lc + ls) * Hc + h) << 8) + d];
  }
  float accl = 0.f;
  for (int t = 0; t < 64; ++t) {
    int ls = l - 63 + t;
    if (ls >= 0) accl += fir_long[hd * 64 + t] * v[((((size_t)b * Lc + ls) * Hc + h) << 8) + d];
  }
  fs[idx] = accs;
  fl[idx] = accl;
}

// ---------------- per-(b,l,h) stats + gate-input assembly (f16, K padded) ----------------
__global__ void __launch_bounds__(256) stats_gate_kernel(
    const float* __restrict__ x, const float* __restrict__ fs, const float* __restrict__ fl,
    const float* __restrict__ delta, const float* __restrict__ v, _Float16* __restrict__ gatein) {
  __shared__ float red[256];
  int r16 = blockIdx.x;
  int c = threadIdx.x;
  int row = r16 >> 2;
  size_t base = (size_t)r16 * DVc + c;
  float vals[4] = { fs[base], fl[base], delta[base], v[base] };
  float mean[4], var[4], am[4], l2[4];
#pragma unroll
  for (int j = 0; j < 4; ++j) {
    float s1 = block_sum(vals[j], red);
    float s2 = block_sum(vals[j] * vals[j], red);
    float sa = block_sum(fabsf(vals[j]), red);
    mean[j] = s1 * (1.f / 256.f);
    var[j]  = s2 * (1.f / 256.f) - mean[j] * mean[j];
    am[j]   = sa * (1.f / 256.f);
    l2[j]   = sqrtf(s2);
  }
  _Float16* grow = gatein + (size_t)r16 * KGATE;
  const float* xr = x + (size_t)row * Dc;
#pragma unroll
  for (int i = 0; i < 4; ++i) grow[c + 256 * i] = (_Float16)xr[c + 256 * i];
  if (c < 16) {
    int br = c >> 2, st = c & 3;
    float sv = (st == 0) ? mean[br] : (st == 1) ? var[br] : (st == 2) ? am[br] : l2[br];
    grow[Dc + c] = (_Float16)sv;
    grow[Dc + 16 + c] = (_Float16)0.f;   // zero pad 1040..1055
  }
}

// ---------------- W2 head + temperature softmax + floor ----------------
__global__ void __launch_bounds__(256) gate_out_kernel(
    const _Float16* __restrict__ h1, const float* __restrict__ W2, const float* __restrict__ b2,
    const float* __restrict__ log_temp, const float* __restrict__ base_bias,
    float* __restrict__ pbuf) {
  __shared__ float red[256];
  int r16 = blockIdx.x;
  int h = r16 & 3;
  int t = threadIdx.x;
  const _Float16* hr = h1 + (size_t)r16 * Dc;
  float part[4] = {0.f, 0.f, 0.f, 0.f};
  for (int c = t; c < Dc; c += 256) {
    float hv = (float)hr[c];
#pragma unroll
    for (int j = 0; j < 4; ++j) part[j] += hv * W2[c * 4 + j];
  }
  float lg[4];
#pragma unroll
  for (int j = 0; j < 4; ++j) lg[j] = block_sum(part[j], red);
  if (t == 0) {
    float temp = log1pf(__expf(log_temp[h])) + 1e-4f;
    float mx = -1e30f;
#pragma unroll
    for (int j = 0; j < 4; ++j) {
      lg[j] = (lg[j] + b2[j] + base_bias[h * 4 + j]) / temp;
      mx = fmaxf(mx, lg[j]);
    }
    float e[4], s = 0.f;
#pragma unroll
    for (int j = 0; j < 4; ++j) { e[j] = __expf(lg[j] - mx); s += e[j]; }
    float p[4];
#pragma unroll
    for (int j = 0; j < 4; ++j) p[j] = e[j] / s;
    p[0] = fmaxf(p[0], 0.05f);
    p[1] = fmaxf(p[1], 0.05f);
    float ps = p[0] + p[1] + p[2] + p[3];
#pragma unroll
    for (int j = 0; j < 4; ++j) pbuf[(size_t)r16 * 4 + j] = p[j] / ps;
  }
}

// ---------------- branch mixing + conv residual + RMSNorm -> f16 (row, h*256+d) ----------------
__global__ void __launch_bounds__(256) mix_kernel(
    const float* __restrict__ fs, const float* __restrict__ fl,
    const float* __restrict__ delta, const float* __restrict__ v,
    const float* __restrict__ pbuf, const float* __restrict__ crl,
    const float* __restrict__ rms_w, _Float16* __restrict__ oh) {
  __shared__ float red[256];
  int r16 = blockIdx.x;
  int c = threadIdx.x;
  int h = r16 & 3;
  int row = r16 >> 2;
  size_t base = (size_t)r16 * DVc + c;
  float p0 = pbuf[(size_t)r16 * 4 + 0], p1 = pbuf[(size_t)r16 * 4 + 1];
  float p2 = pbuf[(size_t)r16 * 4 + 2], p3 = pbuf[(size_t)r16 * 4 + 3];
  float cres = 0.5f * sigmoidf_(crl[h]);
  float o = p0 * fs[base] + p1 * fl[base] + p2 * delta[base] + p3 * v[base] + cres * fs[base];
  float ms = block_sum(o * o, red) * (1.f / 256.f);
  float sc = rsqrtf(ms + 1e-5f);
  oh[(size_t)row * Dc + h * DVc + c] = (_Float16)(o * sc * rms_w[c]);
}

// ---------------- host-side launch ----------------
extern "C" void kernel_launch(void* const* d_in, const int* in_sizes, int n_in,
                              void* d_out, int out_size, void* d_ws, size_t ws_size,
                              hipStream_t stream) {
  (void)in_sizes; (void)n_in; (void)out_size; (void)ws_size;
  const float* x         = (const float*)d_in[0];
  const float* Wq        = (const float*)d_in[1];
  const float* Wk        = (const float*)d_in[2];
  const float* Wv        = (const float*)d_in[3];
  const float* Wb        = (const float*)d_in[4];
  const float* conv_q    = (const float*)d_in[5];
  const float* conv_k    = (const float*)d_in[6];
  const float* conv_v    = (const float*)d_in[7];
  const float* fir_long  = (const float*)d_in[8];
  const float* fir_short = (const float*)d_in[9];
  const float* W1        = (const float*)d_in[10];
  const float* b1        = (const float*)d_in[11];
  const float* W2        = (const float*)d_in[12];
  const float* b2        = (const float*)d_in[13];
  const float* log_temp  = (const float*)d_in[14];
  const float* base_bias = (const float*)d_in[15];
  const float* crl       = (const float*)d_in[16];
  const float* rms_w     = (const float*)d_in[17];
  const float* Wo        = (const float*)d_in[18];

  char* ws = (char*)d_ws;
  size_t off = 0;
  auto carve = [&](size_t bytes) -> char* {
    off = (off + 255) & ~(size_t)255;
    char* p = ws + off;
    off += bytes;
    return p;
  };
  _Float16* xh   = (_Float16*)carve((size_t)Mrows * Dc * 2);
  _Float16* wqh  = (_Float16*)carve((size_t)Dc * Dc * 2);
  _Float16* wkh  = (_Float16*)carve((size_t)Dc * Dc * 2);
  _Float16* wvh  = (_Float16*)carve((size_t)Dc * Dc * 2);
  float*    qp   = (float*)carve((size_t)Mrows * Dc * 4);
  float*    kp   = (float*)carve((size_t)Mrows * Dc * 4);
  float*    vp   = (float*)carve((size_t)Mrows * Dc * 4);
  float*    betab= (float*)carve((size_t)Mrows * Hc * 4);
  _Float16* qn4  = (_Float16*)carve((size_t)Grows * DKc * 2);
  _Float16* kn4  = (_Float16*)carve((size_t)Grows * DKc * 2);
  _Float16* kb4  = (_Float16*)carve((size_t)Grows * DKc * 2);
  _Float16* vb4  = (_Float16*)carve((size_t)Grows * DKc * 2);
  float*    vbl  = (float*)carve((size_t)Grows * DVc * 4);
  float*    dlt  = (float*)carve((size_t)Grows * DVc * 4);
  float*    fsb  = (float*)carve((size_t)Grows * DVc * 4);
  float*    flb  = (float*)carve((size_t)Grows * DVc * 4);
  _Float16* gin  = (_Float16*)carve((size_t)Grows * KGATE * 2);
  _Float16* w1h  = (_Float16*)carve((size_t)KGATE * Dc * 2);
  _Float16* h1h  = (_Float16*)carve((size_t)Grows * Dc * 2);
  float*    pbuf = (float*)carve((size_t)Grows * 4 * 4);
  _Float16* ohb  = (_Float16*)carve((size_t)Mrows * Dc * 2);
  _Float16* woh  = (_Float16*)carve((size_t)Dc * Dc * 2);

  const int nXD = Mrows * Dc;       // 4,194,304
  const int nDD = Dc * Dc;          // 1,048,576

  cvt_f16_kernel<<<nXD / 256, 256, 0, stream>>>(x, xh, nXD);
  cvt_f16_kernel<<<nDD / 256, 256, 0, stream>>>(Wq, wqh, nDD);
  cvt_f16_kernel<<<nDD / 256, 256, 0, stream>>>(Wk, wkh, nDD);
  cvt_f16_kernel<<<nDD / 256, 256, 0, stream>>>(Wv, wvh, nDD);

  int projTiles = (Mrows / 16) * (Dc / 16);          // 16384
  wmma_gemm_kernel<0><<<projTiles / 4, 128, 0, stream>>>(xh, wqh, qp, nullptr,
      Mrows, Dc, Dc, Dc, Dc, Dc);
  wmma_gemm_kernel<0><<<projTiles / 4, 128, 0, stream>>>(xh, wkh, kp, nullptr,
      Mrows, Dc, Dc, Dc, Dc, Dc);
  wmma_gemm_kernel<0><<<projTiles / 4, 128, 0, stream>>>(xh, wvh, vp, nullptr,
      Mrows, Dc, Dc, Dc, Dc, Dc);

  beta_kernel<<<(Mrows * Hc) / 256, 256, 0, stream>>>(x, Wb, betab);
  conv_qkv_kernel<<<Grows, 256, 0, stream>>>(qp, kp, vp, conv_q, conv_k, conv_v,
                                             betab, qn4, kn4, kb4, vb4, vbl);

  delta_kernel<<<Bc * Hc * NSLAB, 256, (int)DELTA_LDS, stream>>>(qn4, kn4, kb4, vb4, dlt);

  fir_kernel<<<(Grows * DVc) / 256, 256, 0, stream>>>(vbl, fir_short, fir_long, fsb, flb);
  stats_gate_kernel<<<Grows, 256, 0, stream>>>(x, fsb, flb, dlt, vbl, gin);

  cvt_pad_kernel<<<(KGATE * Dc) / 256, 256, 0, stream>>>(W1, w1h, Dc + 16, KGATE, Dc);
  int gateTiles = (Grows / 16) * (Dc / 16);          // 65536
  wmma_gemm_kernel<1><<<gateTiles / 4, 128, 0, stream>>>(gin, w1h, h1h, b1,
      Grows, Dc, KGATE, KGATE, Dc, Dc);

  gate_out_kernel<<<Grows, 256, 0, stream>>>(h1h, W2, b2, log_temp, base_bias, pbuf);
  mix_kernel<<<Grows, 256, 0, stream>>>(fsb, flb, dlt, vbl, pbuf, crl, rms_w, ohb);

  cvt_f16_kernel<<<nDD / 256, 256, 0, stream>>>(Wo, woh, nDD);
  wmma_gemm_kernel<0><<<projTiles / 4, 128, 0, stream>>>(ohb, woh, d_out, nullptr,
      Mrows, Dc, Dc, Dc, Dc, Dc);
}
